// ConvNet_MLP_Ja_42803644071990
// MI455X (gfx1250) — matrix operand order repaired and checked by reading, not compiled, not run on hardware
//
#include <hip/hip_runtime.h>
#include <hip/hip_bf16.h>

typedef __attribute__((ext_vector_type(16))) _Float16 v16h;
typedef __attribute__((ext_vector_type(8)))  float    v8f;

// Problem constants
#define NBATCH   4096
#define NAGENT   20
#define NACT     21
#define NID      20
#define NFEAT    32
#define NHID     64
#define NVIEW    210
#define OBS_LD   682          // 210 + 32 + 20*22
#define CONV_OUT 192

// ---------------------------------------------------------------------------
// WMMA fragment loaders (CDNA5 wave32, 16x16x32 f16 layouts per ISA 7.12.2)
// A: 16x32 (M x K chunk).  lane 0-15: M=lane, VGPR i holds K pair
//    (i<4 ? 2i : 16+2(i-4)); lanes 16-31 add +8 to K.  Per lane this is two
//    8-half contiguous runs -> compiler emits 2x b128 loads.
__device__ __forceinline__ v16h load_a16(const _Float16* A, int ld, int rowbase,
                                         int kc, int lane) {
  const int m  = lane & 15;
  const int hi = lane >> 4;
  const _Float16* p = A + (size_t)(rowbase + m) * ld + kc + hi * 8;
  v16h a;
#pragma unroll
  for (int i = 0; i < 8; ++i) {
    const int kb = (i < 4) ? (2 * i) : (16 + 2 * (i - 4));
    a[2 * i]     = p[kb];
    a[2 * i + 1] = p[kb + 1];
  }
  return a;
}

// B fragments are pre-swizzled at prep time into per-lane order:
// frag = 32 lanes x 16 halves contiguous (1KB).  Element e of lane l holds
// W[(kc + (e<8 ? (l>>4)*8+e : 16+(l>>4)*8+e-8)) * ld + nbase + (l&15)].  One aligned v16h load.
__device__ __forceinline__ v16h load_bfrag(const _Float16* F, int frag, int lane) {
  return *(const v16h*)(F + (size_t)frag * 512 + lane * 16);
}

// ---------------------------------------------------------------------------
// K0: swizzle weights into f16 WMMA B-fragment order.
//   w1f: fc1_w (192x64)  -> 6 kc x 4 nt = 24 frags
//   w2f: fc2_w[0:96]x64  -> 3 kc x 4 nt = 12 frags
//   w3f: fc3_w (64x21 -> padded N=32) -> 2 kc x 2 nt = 4 frags
__global__ void prep_weights(const float* __restrict__ fc1_w,
                             const float* __restrict__ fc2_w,
                             const float* __restrict__ fc3_w,
                             _Float16* __restrict__ w1f,
                             _Float16* __restrict__ w2f,
                             _Float16* __restrict__ w3f) {
  const int i = blockIdx.x * blockDim.x + threadIdx.x;
  const int e = i & 15;              // element within lane
  const int l = (i >> 4) & 31;       // lane within fragment
  const int frag = i >> 9;           // 512 elements per fragment
  const int krow = (e < 8) ? ((l >> 4) * 8 + e) : (16 + (l >> 4) * 8 + (e - 8));
  const int ncol = l & 15;
  if (i < 24 * 512) {
    const int kc = (frag >> 2) * 32, nb = (frag & 3) * 16;
    w1f[i] = (_Float16)fc1_w[(kc + krow) * 64 + nb + ncol];
  }
  if (i < 12 * 512) {
    const int kc = (frag >> 2) * 32, nb = (frag & 3) * 16;
    w2f[i] = (_Float16)fc2_w[(kc + krow) * 64 + nb + ncol];
  }
  if (i < 4 * 512) {
    const int kc = (frag >> 1) * 32, nb = (frag & 1) * 16;
    const int k = kc + krow, n = nb + ncol;
    w3f[i] = (n < NACT) ? (_Float16)fc3_w[k * NACT + n] : (_Float16)0.f;
  }
}

// ---------------------------------------------------------------------------
// K1: conv1(5->32, 3x3, 7x6->5x4) + conv2(32->32, 3x3, 5x4->3x2), one block
// per sample.  Weights staged in LDS.  Writes conv-flat (B x 192) f16 and
// feat into xf cols 64..95.  Prefetches the nb section for the agent kernel.
__global__ void conv_kernel(const float* __restrict__ obs,
                            const float* __restrict__ w1, const float* __restrict__ b1,
                            const float* __restrict__ w2, const float* __restrict__ b2,
                            _Float16* __restrict__ xcv, _Float16* __restrict__ xf) {
  __shared__ float view[NVIEW];
  __shared__ float o1[32 * 20];
  __shared__ float w1s[32 * 5 * 9];    // 1440
  __shared__ float w2s[32 * 32 * 9];   // 9216
  __shared__ float b1s[32];
  __shared__ float b2s[32];
  const int b = blockIdx.x;
  const int t = threadIdx.x;
  const float* ob = obs + (size_t)b * OBS_LD;
  for (int i = t; i < NVIEW; i += 256) view[i] = ob[i];
  for (int i = t; i < 1440; i += 256) w1s[i] = w1[i];
  for (int i = t; i < 9216; i += 256) w2s[i] = w2[i];
  if (t < 32) { b1s[t] = b1[t]; b2s[t] = b2[t]; }
  if (t < NFEAT) xf[(size_t)b * 96 + 64 + t] = (_Float16)ob[NVIEW + t];
  // warm L2 for the agent kernel's neighbor-data reads (global_prefetch_b8)
  if (t < NAGENT) __builtin_prefetch(ob + (NVIEW + NFEAT) + t * (1 + NACT), 0, 1);
  __syncthreads();
  // conv1: out[c][y][x], y<5, x<4
  for (int idx = t; idx < 640; idx += 256) {
    const int c = idx / 20, r = idx % 20, y = r >> 2, x = r & 3;
    float acc = b1s[c];
    for (int i = 0; i < 5; ++i)
      for (int dy = 0; dy < 3; ++dy)
        for (int dx = 0; dx < 3; ++dx)
          acc += view[i * 42 + (y + dy) * 6 + (x + dx)] *
                 w1s[((c * 5 + i) * 3 + dy) * 3 + dx];
    o1[idx] = fmaxf(acc, 0.f);
  }
  __syncthreads();
  // conv2: out[c][y][x], y<3, x<2  -> flat col c*6 + y*2 + x
  if (t < CONV_OUT) {
    const int c = t / 6, r = t % 6, y = r >> 1, x = r & 1;
    float acc = b2s[c];
    for (int i = 0; i < 32; ++i)
      for (int dy = 0; dy < 3; ++dy)
        for (int dx = 0; dx < 3; ++dx)
          acc += o1[i * 20 + (y + dy) * 4 + (x + dx)] *
                 w2s[((c * 32 + i) * 3 + dy) * 3 + dx];
    xcv[(size_t)b * CONV_OUT + t] = (_Float16)fmaxf(acc, 0.f);
  }
}

// ---------------------------------------------------------------------------
// K2: x = relu(xcv @ fc1_w + fc1_b); WMMA M=16/block, N=16/wave, K=192.
// 256 blocks x 128 threads (4 waves).
__global__ void fc1_kernel(const _Float16* __restrict__ xcv,
                           const _Float16* __restrict__ w1f,
                           const float* __restrict__ fc1_b,
                           _Float16* __restrict__ xf) {
  const int lane = threadIdx.x & 31;
  const int wv = threadIdx.x >> 5;
  const int rowbase = blockIdx.x * 16;
  v8f c = {};
#pragma unroll
  for (int kt = 0; kt < 6; ++kt) {
    v16h a  = load_a16(xcv, CONV_OUT, rowbase, kt * 32, lane);
    v16h bm = load_bfrag(w1f, kt * 4 + wv, lane);
    c = __builtin_amdgcn_wmma_f32_16x16x32_f16(false, a, false, bm, (short)0, c,
                                               false, false);
  }
  const int n = wv * 16 + (lane & 15);
  const float bias = fc1_b[n];
#pragma unroll
  for (int r = 0; r < 8; ++r) {
    const int m = (lane < 16) ? r : (r + 8);
    xf[(size_t)(rowbase + m) * 96 + n] = (_Float16)fmaxf(c[r] + bias, 0.f);
  }
}

// ---------------------------------------------------------------------------
// K3: base = [x, feat] @ fc2_w[0:96] + fc2_b; WMMA K=96, output f32 (no relu).
__global__ void base_kernel(const _Float16* __restrict__ xf,
                            const _Float16* __restrict__ w2f,
                            const float* __restrict__ fc2_b,
                            float* __restrict__ basev) {
  const int lane = threadIdx.x & 31;
  const int wv = threadIdx.x >> 5;
  const int rowbase = blockIdx.x * 16;
  v8f c = {};
#pragma unroll
  for (int kt = 0; kt < 3; ++kt) {
    v16h a  = load_a16(xf, 96, rowbase, kt * 32, lane);
    v16h bm = load_bfrag(w2f, kt * 4 + wv, lane);
    c = __builtin_amdgcn_wmma_f32_16x16x32_f16(false, a, false, bm, (short)0, c,
                                               false, false);
  }
  const int n = wv * 16 + (lane & 15);
  const float bias = fc2_b[n];
#pragma unroll
  for (int r = 0; r < 8; ++r) {
    const int m = (lane < 16) ? r : (r + 8);
    basev[(size_t)(rowbase + m) * NHID + n] = c[r] + bias;
  }
}

// ---------------------------------------------------------------------------
// K4: per-agent head.  Block = 16 samples x 20 agents = 320 rows, 320 threads
// (10 waves).  Phase 1 builds H rows in LDS (f16); phase 2 WMMA vs fc3
// (K=64 -> 2 chunks, N=21 -> 2 tiles) with bias cnt*fc3_b and agent-sum via
// LDS float atomics; phase 3 stores (B,21).
__global__ void agent_kernel(const float* __restrict__ obs,
                             const float* __restrict__ fc2_w,
                             const float* __restrict__ fc3_b,
                             const float* __restrict__ basev,
                             const _Float16* __restrict__ w3f,
                             float* __restrict__ out) {
  __shared__ _Float16 Hs[320][64];
  __shared__ float w2a[NACT * NHID];   // W2act rows 116..136 of fc2_w
  __shared__ float b3s[24];
  __shared__ float cntv[320];
  __shared__ float oacc[16][24];
  const int t = threadIdx.x;
  const int blk = blockIdx.x;

  for (int i = t; i < NACT * NHID; i += 320) w2a[i] = fc2_w[116 * 64 + i];
  if (t < NACT) b3s[t] = fc3_b[t];
  for (int i = t; i < 16 * 24; i += 320) (&oacc[0][0])[i] = 0.f;
  __syncthreads();

  // ---- phase 1: H rows -------------------------------------------------
  {
    const int bl = t / NAGENT, a = t % NAGENT;
    const size_t b = (size_t)blk * 16 + bl;
    const float* nb = obs + b * OBS_LD + (NVIEW + NFEAT) + a * (1 + NACT);
    const int id = (int)nb[0];
    int actj = -1;
#pragma unroll
    for (int j = 0; j < NACT; ++j)
      if (nb[1 + j] > 0.5f) actj = j;
    const bool valid  = (id >= -1);
    const bool expand = (id >= 0) && (actj < 0);
    const float* bp  = basev + b * NHID;
    const float* idr = (id >= 0) ? (fc2_w + (size_t)(96 + id) * 64) : nullptr;
    cntv[t] = valid ? (expand ? 21.f : 1.f) : 0.f;
    for (int k = 0; k < NHID; ++k) {
      float h = 0.f;
      if (valid) {
        const float vk = bp[k] + (idr ? idr[k] : 0.f);
        if (expand) {
          float acc = 0.f;
#pragma unroll
          for (int j = 0; j < NACT; ++j) acc += fmaxf(vk + w2a[j * 64 + k], 0.f);
          h = acc;                     // sum_j relu(v + W2act_j)
        } else {
          const float av = (actj >= 0) ? w2a[actj * 64 + k] : 0.f;
          h = fmaxf(vk + av, 0.f);
        }
      }
      Hs[t][k] = (_Float16)h;
    }
  }
  __syncthreads();

  // ---- phase 2: WMMA against fc3 --------------------------------------
  {
    const int lane = t & 31;
    const int wv = t >> 5;             // 0..9, each wave owns rows 32w..32w+31
    const v16h b00 = load_bfrag(w3f, 0, lane);  // kc=0,  nt=0
    const v16h b01 = load_bfrag(w3f, 1, lane);  // kc=0,  nt=1
    const v16h b10 = load_bfrag(w3f, 2, lane);  // kc=32, nt=0
    const v16h b11 = load_bfrag(w3f, 3, lane);  // kc=32, nt=1
#pragma unroll
    for (int mt = 0; mt < 2; ++mt) {
      const int rowbase = wv * 32 + mt * 16;
      const v16h a0 = load_a16(&Hs[0][0], 64, rowbase, 0, lane);
      const v16h a1 = load_a16(&Hs[0][0], 64, rowbase, 32, lane);
#pragma unroll
      for (int nt = 0; nt < 2; ++nt) {
        v8f c = {};
        c = __builtin_amdgcn_wmma_f32_16x16x32_f16(false, a0, false,
                                                   nt ? b01 : b00, (short)0, c,
                                                   false, false);
        c = __builtin_amdgcn_wmma_f32_16x16x32_f16(false, a1, false,
                                                   nt ? b11 : b10, (short)0, c,
                                                   false, false);
        const int n = nt * 16 + (lane & 15);
        if (n < NACT) {
          const float bn = b3s[n];
#pragma unroll
          for (int r = 0; r < 8; ++r) {
            const int row = rowbase + ((lane < 16) ? r : (r + 8));
            const float val = c[r] + cntv[row] * bn;  // invalid rows: cnt=0,H=0
            atomicAdd(&oacc[row / NAGENT][n], val);   // ds_add_f32 agent-sum
          }
        }
      }
    }
  }
  __syncthreads();

  for (int i = t; i < 16 * NACT; i += 320) {
    const int bl = i / NACT, n = i % NACT;
    out[((size_t)blk * 16 + bl) * NACT + n] = oacc[bl][n];
  }
}

// ---------------------------------------------------------------------------
extern "C" void kernel_launch(void* const* d_in, const int* in_sizes, int n_in,
                              void* d_out, int out_size, void* d_ws, size_t ws_size,
                              hipStream_t stream) {
  const float* obs   = (const float*)d_in[0];
  const float* c1w   = (const float*)d_in[1];
  const float* c1b   = (const float*)d_in[2];
  const float* c2w   = (const float*)d_in[3];
  const float* c2b   = (const float*)d_in[4];
  const float* fc1_w = (const float*)d_in[5];
  const float* fc1_b = (const float*)d_in[6];
  const float* fc2_w = (const float*)d_in[7];
  const float* fc2_b = (const float*)d_in[8];
  const float* fc3_w = (const float*)d_in[9];
  const float* fc3_b = (const float*)d_in[10];
  float* out = (float*)d_out;

  // workspace layout (all offsets 1KB-aligned for v16h fragment loads)
  char* ws = (char*)d_ws;
  const size_t xcv_off  = 0;
  const size_t xf_off   = xcv_off + (size_t)NBATCH * CONV_OUT * 2;     // f16
  const size_t base_off = xf_off  + (size_t)NBATCH * 96 * 2;           // f16
  const size_t w1f_off  = base_off + (size_t)NBATCH * NHID * 4;        // f32
  const size_t w2f_off  = w1f_off + 24 * 1024;
  const size_t w3f_off  = w2f_off + 12 * 1024;
  _Float16* xcv   = (_Float16*)(ws + xcv_off);
  _Float16* xf    = (_Float16*)(ws + xf_off);
  float*    basev = (float*)(ws + base_off);
  _Float16* w1f   = (_Float16*)(ws + w1f_off);
  _Float16* w2f   = (_Float16*)(ws + w2f_off);
  _Float16* w3f   = (_Float16*)(ws + w3f_off);

  prep_weights<<<48, 256, 0, stream>>>(fc1_w, fc2_w, fc3_w, w1f, w2f, w3f);
  conv_kernel<<<NBATCH, 256, 0, stream>>>(obs, c1w, c1b, c2w, c2b, xcv, xf);
  fc1_kernel<<<NBATCH / 16, 128, 0, stream>>>(xcv, w1f, fc1_b, xf);
  base_kernel<<<NBATCH / 16, 128, 0, stream>>>(xf, w2f, fc2_b, basev);
  agent_kernel<<<NBATCH / 16, 320, 0, stream>>>(obs, fc2_w, fc3_b, basev, w3f, out);
}
